// Model_88630945120366
// MI455X (gfx1250) — matrix operand order repaired and checked by reading, not compiled, not run on hardware
//
#include <hip/hip_runtime.h>
#include <hip/hip_bf16.h>

#define NODES 50000
#define EDGES 800000
#define IDIM  128
#define GRAPHS 64
#define BN_EPS 1e-5f

typedef __attribute__((ext_vector_type(16))) _Float16 v16h;
typedef __attribute__((ext_vector_type(8)))  _Float16 v8h;
typedef __attribute__((ext_vector_type(4)))  _Float16 v4h;
typedef __attribute__((ext_vector_type(8)))  float    v8f;

// ---------------- small device helpers ----------------

__device__ __forceinline__ float eluf(float x) {
    return x > 0.f ? x : expm1f(x);
}

__device__ __forceinline__ float geluf(float x) {
    return 0.5f * x * (1.0f + erff(x * 0.7071067811865476f));
}

// Assemble a v16h from two 8-half LDS chunks
__device__ __forceinline__ v16h load_a16(const _Float16* p0, const _Float16* p1) {
    v8h lo = *(const v8h*)p0;
    v8h hi = *(const v8h*)p1;
    v16h r;
#pragma unroll
    for (int i = 0; i < 8; ++i) { r[i] = lo[i]; r[8 + i] = hi[i]; }
    return r;
}

// CDNA5 async global -> LDS copy of 16 bytes (per-lane addresses, ASYNCcnt)
__device__ __forceinline__ void async_b128(unsigned lds_byte_off, const void* gaddr) {
    asm volatile("global_load_async_to_lds_b128 %0, %1, off"
                 :: "v"(lds_byte_off), "v"(gaddr)
                 : "memory");
}

__device__ __forceinline__ void wait_async0() {
    asm volatile("s_wait_asynccnt 0x0" ::: "memory");
}

__device__ __forceinline__ unsigned lds_off(const void* p) {
    // generic LDS pointer: addr[31:0] is the LDS byte offset (ISA 10.2 aperture calc)
    return (unsigned)(uintptr_t)p;
}

// ---------------- utility kernels ----------------

__global__ void k_convert_w16(const float* __restrict__ W, _Float16* __restrict__ W16, int n) {
    int i = blockIdx.x * blockDim.x + threadIdx.x;
    if (i < n) W16[i] = (_Float16)W[i];
}

__global__ void k_counts(const int* __restrict__ batch, float* __restrict__ counts, int n) {
    int i = blockIdx.x * blockDim.x + threadIdx.x;
    if (i < n) atomicAdd(&counts[batch[i]], 1.0f);
}

// ---------------- CSR-by-destination build (done once, reused by 4 edgeconvs) ----

__global__ void k_hist(const int* __restrict__ dst, int* __restrict__ deg, int n) {
    int e = blockIdx.x * blockDim.x + threadIdx.x;
    if (e < n) atomicAdd(&deg[dst[e]], 1);
}

// single-block chunked exclusive scan: rowptr[0..n] from deg[0..n)
__global__ __launch_bounds__(256) void k_scan_rowptr(const int* __restrict__ deg,
                                                     int* __restrict__ rowptr, int n) {
    __shared__ int sh[256];
    __shared__ int carry;
    if (threadIdx.x == 0) carry = 0;
    __syncthreads();
    for (int base = 0; base < n; base += 256) {
        int i = base + threadIdx.x;
        int v = (i < n) ? deg[i] : 0;
        int c0 = carry;
        sh[threadIdx.x] = v;
        __syncthreads();
        for (int s = 1; s < 256; s <<= 1) {
            int t = (threadIdx.x >= s) ? sh[threadIdx.x - s] : 0;
            __syncthreads();
            sh[threadIdx.x] += t;
            __syncthreads();
        }
        if (i < n) rowptr[i] = c0 + sh[threadIdx.x] - v;   // exclusive
        __syncthreads();
        if (threadIdx.x == 0) carry = c0 + sh[255];
        __syncthreads();
    }
    if (threadIdx.x == 0) rowptr[n] = carry;
}

__global__ void k_copy_int(const int* __restrict__ a, int* __restrict__ b, int n) {
    int i = blockIdx.x * blockDim.x + threadIdx.x;
    if (i < n) b[i] = a[i];
}

__global__ void k_fill_csr(const int* __restrict__ dst, int* __restrict__ cursor,
                           int* __restrict__ eid, int n) {
    int e = blockIdx.x * blockDim.x + threadIdx.x;
    if (e < n) {
        int p = atomicAdd(&cursor[dst[e]], 1);
        eid[p] = e;
    }
}

// ---------------- BN helpers ----------------

// stats: [0..C)=sum, [C..2C)=sumsq  ->  ss: [0..C)=scale, [C..2C)=shift
__global__ void k_finalize_stats(const float* __restrict__ stats,
                                 const float* __restrict__ g, const float* __restrict__ bt,
                                 float* __restrict__ ss, float invcnt, int C) {
    int i = threadIdx.x;
    if (i >= C) return;
    float mean = stats[i] * invcnt;
    float var  = stats[C + i] * invcnt - mean * mean;
    float sc   = g[i] * rsqrtf(var + BN_EPS);
    ss[i]     = sc;
    ss[C + i] = bt[i] - mean * sc;
}

__global__ void k_bn_gelu(const float* __restrict__ T, const float* __restrict__ ss,
                          _Float16* __restrict__ H16, int n) {
    int i = blockIdx.x * blockDim.x + threadIdx.x;
    if (i >= n) return;
    int col = i & (IDIM - 1);
    H16[i] = (_Float16)geluf(T[i] * ss[col] + ss[IDIM + col]);
}

// ---------------- WMMA GEMM kernels ----------------
//
// A-operand (16-bit A 16x32): lanes 0-15: M=lane, K = kt*32+{0..7,16..23}
//                             lanes 16-31: M=lane-16, K = kt*32+{8..15,24..31}
// B-operand: lane l holds col N = l&15, K = same order as A (kt*32+{0..7,16..23} / {8..15,24..31}).
// C/D: col = lane&15; rows: lanes<16 -> M = vgpr, lanes>=16 -> M = 8+vgpr.

// Initial layer: T = X @ W16 + bias (f32 out), accumulate BN stats. X:[Nrows,128] f32.
__global__ __launch_bounds__(256) void k_node_gemm(
    const float* __restrict__ X, const _Float16* __restrict__ W16,
    const float* __restrict__ bias, float* __restrict__ T,
    float* __restrict__ stats, int Nrows)
{
    __shared__ __align__(16) _Float16 lA[32 * 128];
    const int tid = threadIdx.x;
    const int rb  = blockIdx.x * 32;
    {
        const int rl = tid >> 3;
        const int f0 = (tid & 7) * 16;
        const int row = rb + rl;
        _Float16* o = lA + rl * 128 + f0;
        if (row < Nrows) {
            const float4* xp = (const float4*)(X + (size_t)row * IDIM + f0);
#pragma unroll
            for (int q = 0; q < 4; ++q) {
                float4 v = xp[q];
                o[q * 4 + 0] = (_Float16)v.x; o[q * 4 + 1] = (_Float16)v.y;
                o[q * 4 + 2] = (_Float16)v.z; o[q * 4 + 3] = (_Float16)v.w;
            }
        } else {
#pragma unroll
            for (int q = 0; q < 16; ++q) o[q] = (_Float16)0.f;
        }
    }
    __syncthreads();

    const int w    = tid >> 5;
    const int lane = tid & 31;
    const int nt   = w;
    const int colL = lane & 15;
    const int col  = nt * 16 + colL;
    const int hi   = lane >> 4;
    const float bv = bias[col];
    const bool full = (rb + 32 <= Nrows);   // uniform per block
    float s1 = 0.f, s2 = 0.f;

    v8f acc[2] = {};
#pragma unroll
    for (int kt = 0; kt < 4; ++kt) {
        v16h b;
#pragma unroll
        for (int s = 0; s < 16; ++s) b[s] = W16[(size_t)(kt * 32 + hi * 8 + (s & 7) + (s >> 3) * 16) * IDIM + col];
#pragma unroll
        for (int mt = 0; mt < 2; ++mt) {
            const _Float16* ap = lA + (mt * 16 + colL) * 128 + kt * 32 + hi * 8;
            v16h a = load_a16(ap, ap + 16);
            acc[mt] = __builtin_amdgcn_wmma_f32_16x16x32_f16(false, a, false, b,
                                                             (short)0, acc[mt], false, false);
        }
    }
#pragma unroll
    for (int mt = 0; mt < 2; ++mt) {
        const int rowbase = rb + mt * 16 + hi * 8;
        if (full) {
#pragma unroll
            for (int r = 0; r < 8; ++r) {
                float v = acc[mt][r] + bv;
                T[(size_t)(rowbase + r) * IDIM + col] = v;
                s1 += v; s2 += v * v;
            }
        } else {
#pragma unroll
            for (int r = 0; r < 8; ++r) {
                int row = rowbase + r;
                if (row < Nrows) {
                    float v = acc[mt][r] + bv;
                    T[(size_t)row * IDIM + col] = v;
                    s1 += v; s2 += v * v;
                }
            }
        }
    }
    s1 += __shfl_xor(s1, 16);
    s2 += __shfl_xor(s2, 16);
    if (lane < 16) {
        atomicAdd(&stats[col], s1);
        atomicAdd(&stats[IDIM + col], s2);
    }
}

// EdgeConv pass 1: MSGH = [x_i | x_j - x_i] @ W16 + bias (f16 out), BN stats in f32.
// Block: 64 edges (async-gathered f16 rows into LDS), 8 waves x 16-col N-tile, K=256.
__global__ __launch_bounds__(256) void k_edge_gemm(
    const _Float16* __restrict__ H,
    const int* __restrict__ src, const int* __restrict__ dst,
    const _Float16* __restrict__ W16, const float* __restrict__ bias,
    _Float16* __restrict__ MSGH, float* __restrict__ stats)
{
    __shared__ __align__(16) _Float16 lA[64 * 256];   // 32 KB: [edge][ xi(128) | xj->diff(128) ]
    const int tid = threadIdx.x;
    const int eb  = blockIdx.x * 64;

    // ---- async gather: global f16 rows -> LDS (ASYNCcnt) ----
#pragma unroll
    for (int i = 0; i < 2; ++i) {
        const int idx = tid + i * 256;       // 512 (edge, chunk) pairs
        const int el  = idx >> 3;
        const int c   = idx & 7;             // 16-feature chunk
        const int e   = eb + el;
        const int di  = dst[e];
        const int si  = src[e];
        const _Float16* gi = H + (size_t)di * IDIM + c * 16;
        const _Float16* gj = H + (size_t)si * IDIM + c * 16;
        unsigned li = lds_off(lA + el * 256 + c * 16);
        unsigned lj = li + 256;              // +128 halves
        async_b128(li,      gi);
        async_b128(li + 16, gi + 8);
        async_b128(lj,      gj);
        async_b128(lj + 16, gj + 8);
    }
    wait_async0();
    __syncthreads();

    // ---- in-LDS: xj -= xi (packed f16) ----
#pragma unroll
    for (int i = 0; i < 2; ++i) {
        const int idx = tid + i * 256;
        const int el  = idx >> 3;
        const int c   = idx & 7;
        const v8h* pi = (const v8h*)(lA + el * 256 + c * 16);
        v8h*       pj = (v8h*)(lA + el * 256 + 128 + c * 16);
        pj[0] = pj[0] - pi[0];
        pj[1] = pj[1] - pi[1];
    }
    __syncthreads();

    // ---- WMMA: 4 M-tiles x 8 K-steps per wave, B hoisted per K-step ----
    const int w    = tid >> 5;
    const int lane = tid & 31;
    const int nt   = w;
    const int colL = lane & 15;
    const int col  = nt * 16 + colL;
    const int hi   = lane >> 4;
    const float bv = bias[col];

    v8f acc[4] = {};
#pragma unroll
    for (int kt = 0; kt < 8; ++kt) {
        v16h b;
#pragma unroll
        for (int s = 0; s < 16; ++s) b[s] = W16[(size_t)(kt * 32 + hi * 8 + (s & 7) + (s >> 3) * 16) * IDIM + col];
#pragma unroll
        for (int mt = 0; mt < 4; ++mt) {
            const _Float16* ap = lA + (mt * 16 + colL) * 256 + kt * 32 + hi * 8;
            v16h a = load_a16(ap, ap + 16);
            acc[mt] = __builtin_amdgcn_wmma_f32_16x16x32_f16(false, a, false, b,
                                                             (short)0, acc[mt], false, false);
        }
    }

    float s1 = 0.f, s2 = 0.f;
#pragma unroll
    for (int mt = 0; mt < 4; ++mt) {
        const int rowbase = eb + mt * 16 + hi * 8;
#pragma unroll
        for (int r = 0; r < 8; ++r) {
            float v = acc[mt][r] + bv;
            MSGH[(size_t)(rowbase + r) * IDIM + col] = (_Float16)v;
            s1 += v; s2 += v * v;
        }
    }
    s1 += __shfl_xor(s1, 16);
    s2 += __shfl_xor(s2, 16);
    if (lane < 16) {
        atomicAdd(&stats[col], s1);
        atomicAdd(&stats[IDIM + col], s2);
    }
}

// EdgeConv pass 2 via CSR: one wave per node, lane owns 4 columns.
// Streams the node's incoming msg rows (coalesced 256B per row across the wave),
// applies BN-affine + ELU, maxes in registers, writes f16 h row. Zero atomics.
__global__ __launch_bounds__(256) void k_node_max(
    const _Float16* __restrict__ MSGH, const float* __restrict__ ss,
    const int* __restrict__ rowptr, const int* __restrict__ eid,
    _Float16* __restrict__ Hout)
{
    const int node = (blockIdx.x * 256 + threadIdx.x) >> 5;
    const int lane = threadIdx.x & 31;
    if (node >= NODES) return;
    const int beg = rowptr[node];
    const int end = rowptr[node + 1];
    const int c0  = lane * 4;
    float sc[4], sf[4];
#pragma unroll
    for (int j = 0; j < 4; ++j) {
        sc[j] = ss[c0 + j];
        sf[j] = ss[IDIM + c0 + j];
    }
    float mx[4] = {-3.4e38f, -3.4e38f, -3.4e38f, -3.4e38f};
    for (int t = beg; t < end; ++t) {
        const int e = eid[t];
        v4h m = *(const v4h*)(MSGH + (size_t)e * IDIM + c0);
#pragma unroll
        for (int j = 0; j < 4; ++j) {
            float v = (float)m[j] * sc[j] + sf[j];
            mx[j] = fmaxf(mx[j], eluf(v));
        }
    }
    _Float16* o = Hout + (size_t)node * IDIM + c0;
    const bool has = (end > beg);
#pragma unroll
    for (int j = 0; j < 4; ++j) {
        o[j] = has ? (_Float16)mx[j] : (_Float16)0.f;
    }
}

// ---------------- post (Linear I->C, BN, ELU, mean pool) ----------------

__global__ __launch_bounds__(256) void k_post_lin_stats(
    const _Float16* __restrict__ H, const float* __restrict__ W, const float* __restrict__ b,
    int C, float* __restrict__ Y, float* __restrict__ stats, int Nrows)
{
    __shared__ float red[256];
    int n = blockIdx.x * 256 + threadIdx.x;
    bool valid = n < Nrows;
    float acc[2] = {0.f, 0.f};
    if (valid) {
        const v8h* hr = (const v8h*)(H + (size_t)n * IDIM);
        for (int q = 0; q < 16; ++q) {
            v8h hv = hr[q];
#pragma unroll
            for (int j = 0; j < 8; ++j) {
                float h = (float)hv[j];
                for (int c = 0; c < C; ++c) acc[c] += h * W[(q * 8 + j) * C + c];
            }
        }
        for (int c = 0; c < C; ++c) {
            acc[c] += b[c];
            Y[(size_t)n * C + c] = acc[c];
        }
    }
    for (int c = 0; c < C; ++c) {
        for (int p = 0; p < 2; ++p) {
            float v = valid ? (p ? acc[c] * acc[c] : acc[c]) : 0.f;
            red[threadIdx.x] = v;
            __syncthreads();
            for (int s = 128; s > 0; s >>= 1) {
                if (threadIdx.x < s) red[threadIdx.x] += red[threadIdx.x + s];
                __syncthreads();
            }
            if (threadIdx.x == 0) atomicAdd(&stats[p * C + c], red[0]);
            __syncthreads();
        }
    }
}

__global__ __launch_bounds__(256) void k_post_pool(
    const float* __restrict__ Y, const float* __restrict__ ss,
    const int* __restrict__ batch, const float* __restrict__ counts,
    float* __restrict__ out, int C, int col0, int Nrows)
{
    int n = blockIdx.x * 256 + threadIdx.x;
    if (n >= Nrows) return;
    int g = batch[n];
    float cnt = fmaxf(counts[g], 1.0f);
    for (int c = 0; c < C; ++c) {
        float v = eluf(Y[(size_t)n * C + c] * ss[c] + ss[C + c]);
        atomicAdd(&out[g * 2 + col0 + c], v / cnt);
    }
}

// ---------------- host orchestration ----------------

extern "C" void kernel_launch(void* const* d_in, const int* in_sizes, int n_in,
                              void* d_out, int out_size, void* d_ws, size_t ws_size,
                              hipStream_t stream) {
    const float* x       = (const float*)d_in[0];
    const int*   ei      = (const int*)d_in[1];
    const int*   batch   = (const int*)d_in[2];
    const float* init_W  = (const float*)d_in[3];
    const float* init_b  = (const float*)d_in[4];
    const float* init_g  = (const float*)d_in[5];
    const float* init_bt = (const float*)d_in[6];
    const float* mid_W   = (const float*)d_in[7];
    const float* mid_b   = (const float*)d_in[8];
    const float* mid_g   = (const float*)d_in[9];
    const float* mid_bt  = (const float*)d_in[10];
    const float* er_W    = (const float*)d_in[11];
    const float* er_b    = (const float*)d_in[12];
    const float* er_g    = (const float*)d_in[13];
    const float* er_bt   = (const float*)d_in[14];
    const float* pr_W    = (const float*)d_in[15];
    const float* pr_b    = (const float*)d_in[16];
    const float* pr_g    = (const float*)d_in[17];
    const float* pr_bt   = (const float*)d_in[18];
    const float* spost_W = (const float*)d_in[19];
    const float* spost_b = (const float*)d_in[20];
    const float* spost_g = (const float*)d_in[21];
    const float* spost_bt= (const float*)d_in[22];
    const float* erpost_W = (const float*)d_in[23];
    const float* erpost_b = (const float*)d_in[24];
    const float* erpost_g = (const float*)d_in[25];
    const float* erpost_bt= (const float*)d_in[26];
    const float* prpost_W = (const float*)d_in[27];
    const float* prpost_b = (const float*)d_in[28];
    const float* prpost_g = (const float*)d_in[29];
    const float* prpost_bt= (const float*)d_in[30];

    const int* srcI = ei;          // edge_index[0]
    const int* dstI = ei + EDGES;  // edge_index[1]
    float* out = (float*)d_out;

    char* ws = (char*)d_ws;
    size_t off = 0;
    auto carve = [&](size_t bytes) -> void* {
        off = (off + 255) & ~(size_t)255;
        void* p = ws + off;
        off += bytes;
        return p;
    };
    _Float16* h16A   = (_Float16*)carve((size_t)NODES * IDIM * 2);
    _Float16* h16B   = (_Float16*)carve((size_t)NODES * IDIM * 2);
    float*    tmpf   = (float*)carve((size_t)NODES * IDIM * 4);   // init-layer pre-BN T
    _Float16* msgh   = (_Float16*)carve((size_t)EDGES * IDIM * 2);
    _Float16* w16    = (_Float16*)carve((size_t)256 * 128 * 2);
    float*    stats  = (float*)carve(2 * IDIM * 4);
    float*    ss     = (float*)carve(2 * IDIM * 4);
    float*    py     = (float*)carve((size_t)NODES * 2 * 4);
    float*    pstats = (float*)carve(4 * 4);
    float*    pss    = (float*)carve(4 * 4);
    float*    counts = (float*)carve(GRAPHS * 4);
    int*      deg    = (int*)carve((size_t)NODES * 4);
    int*      rowptr = (int*)carve((size_t)(NODES + 1) * 4);
    int*      cursor = (int*)carve((size_t)NODES * 4);
    int*      eid    = (int*)carve((size_t)EDGES * 4);

    const int NB256 = (NODES + 255) / 256;   // node-parallel blocks
    const int NHB   = (NODES * IDIM) / 256;  // per-element blocks over h
    const int EB256 = EDGES / 256;           // edge-parallel blocks (exact)

    hipMemsetAsync(d_out, 0, GRAPHS * 2 * sizeof(float), stream);
    hipMemsetAsync(counts, 0, GRAPHS * sizeof(float), stream);
    k_counts<<<NB256, 256, 0, stream>>>(batch, counts, NODES);

    // ---- build CSR by destination (once; reused by all 4 edgeconvs) ----
    hipMemsetAsync(deg, 0, (size_t)NODES * 4, stream);
    k_hist<<<EB256, 256, 0, stream>>>(dstI, deg, EDGES);
    k_scan_rowptr<<<1, 256, 0, stream>>>(deg, rowptr, NODES);
    k_copy_int<<<NB256, 256, 0, stream>>>(rowptr, cursor, NODES);
    k_fill_csr<<<EB256, 256, 0, stream>>>(dstI, cursor, eid, EDGES);

    auto do_post = [&](const _Float16* H, const float* W, const float* b,
                       const float* g, const float* bt, int C, int col0) {
        hipMemsetAsync(pstats, 0, 2 * C * sizeof(float), stream);
        k_post_lin_stats<<<NB256, 256, 0, stream>>>(H, W, b, C, py, pstats, NODES);
        k_finalize_stats<<<1, C, 0, stream>>>(pstats, g, bt, pss, 1.0f / NODES, C);
        k_post_pool<<<NB256, 256, 0, stream>>>(py, pss, batch, counts, out, C, col0, NODES);
    };

    auto do_edgeconv = [&](const _Float16* Hin, _Float16* Hout, const float* W, const float* b,
                           const float* g, const float* bt) {
        k_convert_w16<<<(256 * 128) / 256, 256, 0, stream>>>(W, w16, 256 * 128);
        hipMemsetAsync(stats, 0, 2 * IDIM * sizeof(float), stream);
        k_edge_gemm<<<EDGES / 64, 256, 0, stream>>>(Hin, srcI, dstI, w16, b, msgh, stats);
        k_finalize_stats<<<1, IDIM, 0, stream>>>(stats, g, bt, ss, 1.0f / EDGES, IDIM);
        k_node_max<<<(NODES + 7) / 8, 256, 0, stream>>>(msgh, ss, rowptr, eid, Hout);
    };

    // ---- initial subnet: h = gelu(BN(x @ init_W + init_b)) ----
    k_convert_w16<<<(128 * 128) / 256, 256, 0, stream>>>(init_W, w16, 128 * 128);
    hipMemsetAsync(stats, 0, 2 * IDIM * sizeof(float), stream);
    k_node_gemm<<<(NODES + 31) / 32, 256, 0, stream>>>(x, w16, init_b, tmpf, stats, NODES);
    k_finalize_stats<<<1, IDIM, 0, stream>>>(stats, init_g, init_bt, ss, 1.0f / NODES, IDIM);
    k_bn_gelu<<<NHB, 256, 0, stream>>>(tmpf, ss, h16A, NODES * IDIM);
    do_post(h16A, spost_W, spost_b, spost_g, spost_bt, 2, 0);

    // ---- 2 middle EdgeConvs ----
    do_edgeconv(h16A, h16B, mid_W, mid_b, mid_g, mid_bt);
    do_post(h16B, spost_W + 256, spost_b + 2, spost_g + 2, spost_bt + 2, 2, 0);

    do_edgeconv(h16B, h16A, mid_W + 256 * 128, mid_b + 128, mid_g + 128, mid_bt + 128);
    do_post(h16A, spost_W + 512, spost_b + 4, spost_g + 4, spost_bt + 4, 2, 0);

    // ---- er / pr branches (both read h16A; sequential reuse of h16B) ----
    do_edgeconv(h16A, h16B, er_W, er_b, er_g, er_bt);
    do_post(h16B, erpost_W, erpost_b, erpost_g, erpost_bt, 1, 0);

    do_edgeconv(h16A, h16B, pr_W, pr_b, pr_g, pr_bt);
    do_post(h16B, prpost_W, prpost_b, prpost_g, prpost_bt, 1, 1);
}
